// SharedPPOSoftModularizedMLP_66383014527285
// MI455X (gfx1250) — matrix operand order repaired and checked by reading, not compiled, not run on hardware
//
#include <hip/hip_runtime.h>

// ---------------------------------------------------------------------------
// Types for CDNA5 WMMA
// ---------------------------------------------------------------------------
typedef __attribute__((ext_vector_type(16))) __bf16 v16bf;
typedef __attribute__((ext_vector_type(8)))  float  v8f;

union FragBF16 {
    v16bf v;
    uint4 q[2];
};

// fp32 -> bf16 (round to nearest even) stored as raw ushort
__device__ __forceinline__ unsigned short f2bf(float f) {
    unsigned int u = __float_as_uint(f);
    unsigned int r = u + 0x7FFFu + ((u >> 16) & 1u);
    return (unsigned short)(r >> 16);
}
__device__ __forceinline__ float bf2f(unsigned short h) {
    return __uint_as_float(((unsigned int)h) << 16);
}

// ---------------------------------------------------------------------------
// CDNA5 async global->LDS copy (ASYNCcnt-tracked), 16 bytes per lane.
// ldsOff: wave-relative LDS byte address in a VGPR; g: 64-bit global address.
// ---------------------------------------------------------------------------
__device__ __forceinline__ void async_copy_b128(unsigned ldsOff, const void* g) {
    asm volatile("global_load_async_to_lds_b128 %0, %1, off"
                 :: "v"(ldsOff), "v"((unsigned long long)g)
                 : "memory");
}
__device__ __forceinline__ void wait_async0() {
    asm volatile("s_wait_asynccnt 0x0" ::: "memory");
}
__device__ __forceinline__ unsigned lds_off(const void* p) {
    // generic pointer to LDS: low 32 bits are the wave-relative LDS address
    return (unsigned)(unsigned long long)(p);
}

// ---------------------------------------------------------------------------
// Problem constants
// ---------------------------------------------------------------------------
static constexpr int BATCH   = 8192;
static constexpr int OBS     = 128;
static constexpr int TASK    = 64;
static constexpr int DIM     = 1024;
static constexpr int NE      = 8;      // experts
static constexpr int E2      = 64;     // NE*NE

// GEMM tiling
static constexpr int BM = 128;
static constexpr int BN = 128;
static constexpr int BK = 32;

// ---------------------------------------------------------------------------
// Tiled bf16 WMMA GEMM:  C[M,Nout] = act( A[M,K](bf16) @ Wt[Npad,K]^T + bias )
// Wt is stored N-major (row n holds K contiguous bf16) — i.e. W transposed.
// Double-buffered LDS fed by GLOBAL_LOAD_ASYNC_TO_LDS_B128; one barrier per
// K-step, async copies overlapped with the 8 WMMAs per wave per step.
// Optional fp32 and/or bf16 outputs. Batched over blockIdx.z via strides.
// ---------------------------------------------------------------------------
__global__ __launch_bounds__(256, 2)
void gemm_bf16_wmma(const unsigned short* __restrict__ A, int lda,
                    const unsigned short* __restrict__ Wt,
                    const float* __restrict__ bias, int biasStride,
                    float* __restrict__ Cf, unsigned short* __restrict__ Cbf,
                    int ldc, int M, int Nout, int K, int relu,
                    long strideA, long strideW, long strideC)
{
    __shared__ unsigned short lA[2][BM * BK];
    __shared__ unsigned short lB[2][BN * BK];

    const int bz = blockIdx.z;
    A  += (long)bz * strideA;
    Wt += (long)bz * strideW;
    const long coff = (long)bz * strideC;
    const float* bptr = bias ? (bias + (long)bz * biasStride) : nullptr;

    const int m0 = blockIdx.x * BM;
    const int n0 = blockIdx.y * BN;
    const int t    = threadIdx.x;
    const int wave = t >> 5;
    const int lane = t & 31;
    const int wr = wave >> 1;          // 0..3  (row group of 32)
    const int wc = wave & 1;           // 0..1  (col group of 64)
    const int lhalf = lane >> 4;       // 0/1
    const int l16   = lane & 15;

    // staging map: thread t async-copies 16 bf16 of one A row and one B row
    const int lrow = t >> 1;           // 0..127
    const int lcol = (t & 1) * 16;     // 0 or 16

    const unsigned short* gA = A  + (long)(m0 + lrow) * lda + lcol;
    const unsigned short* gB = Wt + (long)(n0 + lrow) * K   + lcol;
    const unsigned laOff[2] = { lds_off(&lA[0][lrow * BK + lcol]),
                                lds_off(&lA[1][lrow * BK + lcol]) };
    const unsigned lbOff[2] = { lds_off(&lB[0][lrow * BK + lcol]),
                                lds_off(&lB[1][lrow * BK + lcol]) };

    auto issue = [&](int buf, int k0) {
        async_copy_b128(laOff[buf],      gA + k0);
        async_copy_b128(laOff[buf] + 16, gA + k0 + 8);
        async_copy_b128(lbOff[buf],      gB + k0);
        async_copy_b128(lbOff[buf] + 16, gB + k0 + 8);
    };

    v8f acc[2][4] = {};

    const int nk = K / BK;
    issue(0, 0);
    wait_async0();
    __syncthreads();

    for (int ks = 0; ks < nk; ++ks) {
        const int cur = ks & 1;
        if (ks + 1 < nk) issue(cur ^ 1, (ks + 1) * BK);

        const unsigned short* curA = lA[cur];
        const unsigned short* curB = lB[cur];

        // B fragments: lane holds N = l16, halves h -> K = 16*lhalf + h
        FragBF16 fb[4];
#pragma unroll
        for (int nt = 0; nt < 4; ++nt) {
            const int n = wc * 64 + nt * 16 + l16;
            const uint4* p = (const uint4*)(&curB[n * BK + lhalf * 16]);
            fb[nt].q[0] = p[0];
            fb[nt].q[1] = p[1];
        }

#pragma unroll
        for (int mt = 0; mt < 2; ++mt) {
            // A fragment: lane row M = l16, K = 16*half .. +15 contiguous
            const int m = wr * 32 + mt * 16 + l16;
            FragBF16 fa;
            fa.q[0] = *(const uint4*)(&curA[m * BK + lhalf * 16]);      // K = 16*half .. +7
            fa.q[1] = *(const uint4*)(&curA[m * BK + lhalf * 16 + 8]);  // K = 16*half+8 .. +15
#pragma unroll
            for (int nt = 0; nt < 4; ++nt) {
                acc[mt][nt] = __builtin_amdgcn_wmma_f32_16x16x32_bf16(
                    false, fa.v, false, fb[nt].v,
                    (short)0, acc[mt][nt], false, false);
            }
        }

        wait_async0();     // next tiles resident in LDS
        __syncthreads();   // all waves done reading current tiles
    }

    // epilogue: C/D layout -> lane N = l16, VGPR v -> M = v + 8*lhalf
#pragma unroll
    for (int mt = 0; mt < 2; ++mt) {
#pragma unroll
        for (int nt = 0; nt < 4; ++nt) {
            const int n = n0 + wc * 64 + nt * 16 + l16;
            if (n < Nout) {
                const float bv = bptr ? bptr[n] : 0.0f;
#pragma unroll
                for (int v = 0; v < 8; ++v) {
                    const int m = m0 + wr * 32 + mt * 16 + v + 8 * lhalf;
                    float val = acc[mt][nt][v] + bv;
                    if (relu) val = fmaxf(val, 0.0f);
                    const long off = coff + (long)m * ldc + n;
                    if (Cf)  Cf[off]  = val;
                    if (Cbf) Cbf[off] = f2bf(val);
                }
            }
        }
    }
}

// ---------------------------------------------------------------------------
// Weight convert: W[K,N] fp32 -> Wt[Npad,K] bf16 (transposed, zero padded),
// batched over nmats.
// ---------------------------------------------------------------------------
__global__ void cvt_wt_kernel(const float* __restrict__ W,
                              unsigned short* __restrict__ Wt,
                              int K, int N, int Npad, int nmats, long wStride)
{
    const long per = (long)Npad * K;
    const long tot = per * nmats;
    long tid = (long)blockIdx.x * blockDim.x + threadIdx.x;
    if (tid >= tot) return;
    const int  mat = (int)(tid / per);
    const long r   = tid % per;
    const int  n   = (int)(r / K);
    const int  k   = (int)(r % K);
    float v = (n < N) ? W[(long)mat * wStride + (long)k * N + n] : 0.0f;
    Wt[tid] = f2bf(v);
}

__global__ void pad_bias_kernel(const float* __restrict__ b, float* __restrict__ bp,
                                int N, int Npad, int nmats, int bStride)
{
    int tid = blockIdx.x * blockDim.x + threadIdx.x;
    if (tid >= Npad * nmats) return;
    const int mat = tid / Npad;
    const int n   = tid % Npad;
    bp[tid] = (n < N) ? b[mat * bStride + n] : 0.0f;
}

// x[B,192] fp32 -> obs_bf[B,128], task_bf[B,64]
__global__ void cvt_x_kernel(const float* __restrict__ x,
                             unsigned short* __restrict__ obs,
                             unsigned short* __restrict__ task)
{
    long tid = (long)blockIdx.x * blockDim.x + threadIdx.x;
    if (tid >= (long)BATCH * (OBS + TASK)) return;
    const int b = (int)(tid / (OBS + TASK));
    const int c = (int)(tid % (OBS + TASK));
    unsigned short v = f2bf(x[tid]);
    if (c < OBS) obs[(long)b * OBS + c] = v;
    else         task[(long)b * TASK + (c - OBS)] = v;
}

// r_in = f_obs * z ; rrelu_bf = bf16(relu(r_in))
__global__ void rin_kernel(const float* __restrict__ f, const float* __restrict__ z,
                           float* __restrict__ r, unsigned short* __restrict__ rr, long n)
{
    long tid = (long)blockIdx.x * blockDim.x + threadIdx.x;
    if (tid >= n) return;
    float v = f[tid] * z[tid];
    r[tid]  = v;
    rr[tid] = f2bf(fmaxf(v, 0.0f));
}

// t_bf = bf16(relu(u * r_in))
__global__ void ur_kernel(const float* __restrict__ u, const float* __restrict__ r,
                          unsigned short* __restrict__ o, long n)
{
    long tid = (long)blockIdx.x * blockDim.x + threadIdx.x;
    if (tid >= n) return;
    o[tid] = f2bf(fmaxf(u[tid] * r[tid], 0.0f));
}

// softmax over contiguous groups of 8
__global__ void softmax8_kernel(const float* __restrict__ p, float* __restrict__ pr, int rows)
{
    int tid = blockIdx.x * blockDim.x + threadIdx.x;
    if (tid >= rows) return;
    const float* q = p  + (long)tid * 8;
    float*       o = pr + (long)tid * 8;
    float mx = q[0];
#pragma unroll
    for (int j = 1; j < 8; ++j) mx = fmaxf(mx, q[j]);
    float e[8], s = 0.0f;
#pragma unroll
    for (int j = 0; j < 8; ++j) { e[j] = __expf(q[j] - mx); s += e[j]; }
    const float inv = 1.0f / s;
#pragma unroll
    for (int j = 0; j < 8; ++j) o[j] = e[j] * inv;
}

// xx[i,b,d] = sum_j probs[b, i*8+j] * he[j,b,d]
__global__ void mix_kernel(const float* __restrict__ probs,
                           const unsigned short* __restrict__ he,
                           unsigned short* __restrict__ xx)
{
    long tid = (long)blockIdx.x * blockDim.x + threadIdx.x;
    const long tot = (long)BATCH * DIM;
    if (tid >= tot) return;
    const int b = (int)(tid / DIM);
    float h[NE];
#pragma unroll
    for (int j = 0; j < NE; ++j) h[j] = bf2f(he[(long)j * tot + tid]);
    const float* pr = probs + (long)b * E2;
#pragma unroll
    for (int i = 0; i < NE; ++i) {
        float s = 0.0f;
#pragma unroll
        for (int j = 0; j < NE; ++j) s += pr[i * 8 + j] * h[j];
        xx[(long)i * tot + tid] = f2bf(s);
    }
}

// mu[b,o] = sum_e (xx[e,b,:].aW[e,:,o] + ab[e,o]); values[b] = sum_e (xx[e,b,:].cW[e,:,0] + cb[e])
__global__ __launch_bounds__(256)
void heads_kernel(const unsigned short* __restrict__ xx,
                  const float* __restrict__ aW, const float* __restrict__ ab,
                  const float* __restrict__ cW, const float* __restrict__ cb,
                  float* __restrict__ out)
{
    const int b    = blockIdx.x;
    const int e    = threadIdx.x >> 5;
    const int lane = threadIdx.x & 31;
    const long base = ((long)e * BATCH + b) * DIM;

    float acc[9] = {0, 0, 0, 0, 0, 0, 0, 0, 0};
    for (int d = lane; d < DIM; d += 32) {
        const float xv = bf2f(xx[base + d]);
        const float* aw = aW + ((long)e * DIM + d) * 8;
#pragma unroll
        for (int o = 0; o < 8; ++o) acc[o] += xv * aw[o];
        acc[8] += xv * cW[(long)e * DIM + d];
    }
#pragma unroll
    for (int off = 16; off > 0; off >>= 1) {
#pragma unroll
        for (int o = 0; o < 9; ++o) acc[o] += __shfl_down(acc[o], off, 32);
    }
    __shared__ float sm[NE][9];
    if (lane == 0) {
#pragma unroll
        for (int o = 0; o < 9; ++o) sm[e][o] = acc[o];
    }
    __syncthreads();
    if (threadIdx.x < 9) {
        const int o = threadIdx.x;
        float s = 0.0f;
        if (o < 8) {
            for (int k = 0; k < NE; ++k) s += sm[k][o] + ab[k * 8 + o];
            out[(long)b * 8 + o] = s;
        } else {
            for (int k = 0; k < NE; ++k) s += sm[k][8] + cb[k];
            out[(long)BATCH * 8 + b] = s;
        }
    }
}

// ---------------------------------------------------------------------------
// Host launcher
// ---------------------------------------------------------------------------
static inline long cdiv(long a, long b) { return (a + b - 1) / b; }

extern "C" void kernel_launch(void* const* d_in, const int* in_sizes, int n_in,
                              void* d_out, int out_size, void* d_ws, size_t ws_size,
                              hipStream_t stream)
{
    (void)in_sizes; (void)n_in; (void)out_size; (void)ws_size;

    const float* x    = (const float*)d_in[0];
    const float* sW1  = (const float*)d_in[1];
    const float* sb1  = (const float*)d_in[2];
    const float* sW2  = (const float*)d_in[3];
    const float* sb2  = (const float*)d_in[4];
    const float* sW3  = (const float*)d_in[5];
    const float* sb3  = (const float*)d_in[6];
    const float* tW   = (const float*)d_in[7];
    const float* tb   = (const float*)d_in[8];
    const float* rdW  = (const float*)d_in[9];   // (3,1024,64)
    const float* rdb  = (const float*)d_in[10];  // (3,64)
    const float* ruW  = (const float*)d_in[11];  // (2,64,1024)
    const float* rub  = (const float*)d_in[12];  // (2,1024)
    const float* eW   = (const float*)d_in[13];  // (3,8,1024,1024)
    const float* eb   = (const float*)d_in[14];  // (3,8,1,1024)
    const float* aW   = (const float*)d_in[15];
    const float* ab   = (const float*)d_in[16];
    const float* cW   = (const float*)d_in[17];
    const float* cb   = (const float*)d_in[18];
    float* out = (float*)d_out;

    // bump allocator over workspace
    char* wsp = (char*)d_ws;
    auto alloc = [&](size_t bytes) -> void* {
        void* p = (void*)wsp;
        wsp += (bytes + 255) & ~(size_t)255;
        return p;
    };

    const long BD  = (long)BATCH * DIM;        // 8M elements
    const long EBD = (long)NE * BD;

    unsigned short* obs_bf  = (unsigned short*)alloc((size_t)BATCH * OBS * 2);
    unsigned short* task_bf = (unsigned short*)alloc((size_t)BATCH * TASK * 2);
    unsigned short* sW1t    = (unsigned short*)alloc((size_t)512 * 128 * 2);
    unsigned short* sW2t    = (unsigned short*)alloc((size_t)512 * 512 * 2);
    unsigned short* sW3t    = (unsigned short*)alloc((size_t)1024 * 512 * 2);
    unsigned short* tWt     = (unsigned short*)alloc((size_t)1024 * 64 * 2);
    unsigned short* rdWt    = (unsigned short*)alloc((size_t)2 * 128 * 1024 * 2); // layers 0,1, N padded 64->128
    unsigned short* ruW0t   = (unsigned short*)alloc((size_t)1024 * 64 * 2);
    unsigned short* eW0t    = (unsigned short*)alloc((size_t)NE * DIM * DIM * 2);
    unsigned short* eW1t    = (unsigned short*)alloc((size_t)NE * DIM * DIM * 2);
    float*          rdb_pad = (float*)alloc((size_t)2 * 128 * 4);
    unsigned short* h1_bf   = (unsigned short*)alloc((size_t)BATCH * 512 * 2);
    unsigned short* h2_bf   = (unsigned short*)alloc((size_t)BATCH * 512 * 2);
    float*          f_obs   = (float*)alloc((size_t)BD * 4);
    unsigned short* fobs_bf = (unsigned short*)alloc((size_t)BD * 2);
    float*          z_u1    = (float*)alloc((size_t)BD * 4);       // z, then reused as u1
    float*          r_in    = (float*)alloc((size_t)BD * 4);
    unsigned short* rr_t1   = (unsigned short*)alloc((size_t)BD * 2); // relu(r_in), then t1
    float*          p0      = (float*)alloc((size_t)BATCH * E2 * 4);
    unsigned short* p0_bf   = (unsigned short*)alloc((size_t)BATCH * E2 * 2);
    float*          p1      = (float*)alloc((size_t)BATCH * E2 * 4);
    float*          probs0  = (float*)alloc((size_t)BATCH * E2 * 4);
    float*          probs1  = (float*)alloc((size_t)BATCH * E2 * 4);
    unsigned short* he_bf   = (unsigned short*)alloc((size_t)EBD * 2);
    unsigned short* xx_bf   = (unsigned short*)alloc((size_t)EBD * 2);

    const int TB = 256;

    // --- convert inputs / weights to bf16 (transposed, padded) ---
    cvt_x_kernel<<<cdiv((long)BATCH * 192, TB), TB, 0, stream>>>(x, obs_bf, task_bf);
    cvt_wt_kernel<<<cdiv((long)512 * 128, TB), TB, 0, stream>>>(sW1, sW1t, 128, 512, 512, 1, 0);
    cvt_wt_kernel<<<cdiv((long)512 * 512, TB), TB, 0, stream>>>(sW2, sW2t, 512, 512, 512, 1, 0);
    cvt_wt_kernel<<<cdiv((long)1024 * 512, TB), TB, 0, stream>>>(sW3, sW3t, 512, 1024, 1024, 1, 0);
    cvt_wt_kernel<<<cdiv((long)1024 * 64, TB), TB, 0, stream>>>(tW, tWt, 64, 1024, 1024, 1, 0);
    cvt_wt_kernel<<<cdiv((long)2 * 128 * 1024, TB), TB, 0, stream>>>(rdW, rdWt, 1024, 64, 128, 2, (long)1024 * 64);
    cvt_wt_kernel<<<cdiv((long)1024 * 64, TB), TB, 0, stream>>>(ruW, ruW0t, 64, 1024, 1024, 1, 0);
    cvt_wt_kernel<<<cdiv((long)NE * DIM * DIM, TB), TB, 0, stream>>>(eW, eW0t, DIM, DIM, DIM, NE, (long)DIM * DIM);
    cvt_wt_kernel<<<cdiv((long)NE * DIM * DIM, TB), TB, 0, stream>>>(eW + (long)NE * DIM * DIM, eW1t, DIM, DIM, DIM, NE, (long)DIM * DIM);
    pad_bias_kernel<<<cdiv(2 * 128, TB), TB, 0, stream>>>(rdb, rdb_pad, 64, 128, 2, 64);

    const dim3 blk(256);

    // --- shared MLP ---
    gemm_bf16_wmma<<<dim3(BATCH / BM, 512 / BN, 1), blk, 0, stream>>>(
        obs_bf, 128, sW1t, sb1, 0, nullptr, h1_bf, 512, BATCH, 512, 128, 1, 0, 0, 0);
    gemm_bf16_wmma<<<dim3(BATCH / BM, 512 / BN, 1), blk, 0, stream>>>(
        h1_bf, 512, sW2t, sb2, 0, nullptr, h2_bf, 512, BATCH, 512, 512, 1, 0, 0, 0);
    gemm_bf16_wmma<<<dim3(BATCH / BM, DIM / BN, 1), blk, 0, stream>>>(
        h2_bf, 512, sW3t, sb3, 0, f_obs, fobs_bf, DIM, BATCH, DIM, 512, 0, 0, 0, 0);

    // --- task embed + routing input ---
    gemm_bf16_wmma<<<dim3(BATCH / BM, DIM / BN, 1), blk, 0, stream>>>(
        task_bf, 64, tWt, tb, 0, z_u1, nullptr, DIM, BATCH, DIM, 64, 1, 0, 0, 0);
    rin_kernel<<<cdiv(BD, TB), TB, 0, stream>>>(f_obs, z_u1, r_in, rr_t1, BD);

    // --- routing: p0, u1, p1 (p2 branch is dead code w.r.t. outputs) ---
    gemm_bf16_wmma<<<dim3(BATCH / BM, 1, 1), blk, 0, stream>>>(
        rr_t1, DIM, rdWt, rdb_pad, 0, p0, p0_bf, E2, BATCH, E2, DIM, 0, 0, 0, 0);
    gemm_bf16_wmma<<<dim3(BATCH / BM, DIM / BN, 1), blk, 0, stream>>>(
        p0_bf, E2, ruW0t, rub, 0, z_u1, nullptr, DIM, BATCH, DIM, E2, 0, 0, 0, 0);
    ur_kernel<<<cdiv(BD, TB), TB, 0, stream>>>(z_u1, r_in, rr_t1, BD);
    gemm_bf16_wmma<<<dim3(BATCH / BM, 1, 1), blk, 0, stream>>>(
        rr_t1, DIM, rdWt + (long)128 * 1024, rdb_pad + 128, 0, p1, nullptr, E2, BATCH, E2, DIM, 0, 0, 0, 0);

    softmax8_kernel<<<cdiv(BATCH * NE, TB), TB, 0, stream>>>(p0, probs0, BATCH * NE);
    softmax8_kernel<<<cdiv(BATCH * NE, TB), TB, 0, stream>>>(p1, probs1, BATCH * NE);

    // --- MoE layer 0: he = relu(f_obs @ eW0[e] + eb0[e]), batched over experts ---
    gemm_bf16_wmma<<<dim3(BATCH / BM, DIM / BN, NE), blk, 0, stream>>>(
        fobs_bf, DIM, eW0t, eb, DIM, nullptr, he_bf, DIM, BATCH, DIM, DIM, 1,
        0, (long)DIM * DIM, BD);
    mix_kernel<<<cdiv(BD, TB), TB, 0, stream>>>(probs0, he_bf, xx_bf);

    // --- MoE layer 1 ---
    gemm_bf16_wmma<<<dim3(BATCH / BM, DIM / BN, NE), blk, 0, stream>>>(
        xx_bf, DIM, eW1t, eb + (long)NE * DIM, DIM, nullptr, he_bf, DIM, BATCH, DIM, DIM, 1,
        BD, (long)DIM * DIM, BD);
    mix_kernel<<<cdiv(BD, TB), TB, 0, stream>>>(probs1, he_bf, xx_bf);

    // --- heads ---
    heads_kernel<<<dim3(BATCH), dim3(256), 0, stream>>>(xx_bf, aW, ab, cW, cb, out);
}
